// Quantizer_59081570125221
// MI455X (gfx1250) — matrix operand order repaired and checked by reading, not compiled, not run on hardware
//
#include <hip/hip_runtime.h>
#include <hip/hip_bf16.h>

typedef __attribute__((ext_vector_type(16))) _Float16 v16h;
typedef __attribute__((ext_vector_type(4)))  _Float16 v4h;
typedef __attribute__((ext_vector_type(8)))  float    v8f;

#define C_DIM   64
#define NE      512
#define PIX_WG  256   // pixels per workgroup
#define NBLK    1024  // 262144 / 256

__global__ __launch_bounds__(256) void vq_argmin_gather(
    const float* __restrict__ x,   // [64,64,64,64] = [B,C,H,W]
    const float* __restrict__ e,   // [512,64]
    float* __restrict__ out)       // [B,C,H,W]
{
    __shared__ _Float16 eL[NE * C_DIM];   // 64 KB codebook in f16
    __shared__ float    nrm[NE];          // ||e_n||^2 in f32
    __shared__ int      idxL[PIX_WG];     // winning index per pixel

    const int t = threadIdx.x;

    // ---------------- stage codebook: f32 -> f16 LDS + exact f32 norms ----
    {
        #pragma unroll
        for (int rr = 0; rr < 2; ++rr) {
            const int r = t * 2 + rr;                 // rows 0..511
            const float* er = e + r * C_DIM;
            float s = 0.f;
            #pragma unroll
            for (int k = 0; k < C_DIM; k += 4) {
                float4 v = *(const float4*)(er + k);
                s += v.x * v.x + v.y * v.y + v.z * v.z + v.w * v.w;
                v4h h = { (_Float16)v.x, (_Float16)v.y, (_Float16)v.z, (_Float16)v.w };
                *(v4h*)(eL + r * C_DIM + k) = h;      // 8B-aligned ds_store
            }
            nrm[r] = s;
        }
    }
    __syncthreads();

    // ---------------- per-wave GEMM + argmin --------------------------------
    const int wave = t >> 5;
    const int lane = t & 31;
    const int l16  = lane & 15;
    const bool hi  = lane >= 16;
    const int koff = hi ? 16 : 0;

    const int basePix = blockIdx.x * PIX_WG + wave * 32;   // 32 pixels per wave

    // A fragments: [rowtile][kchunk], contiguous k = 16*h + j within each 32-k chunk
    v16h afrag[2][2];
    #pragma unroll
    for (int rt = 0; rt < 2; ++rt) {
        const int p   = basePix + rt * 16 + l16;           // this lane's row/pixel
        const int b   = p >> 12;
        const int rem = p & 4095;
        const float* px = x + (size_t)b * 262144 + rem;    // channel c at px[c*4096]
        #pragma unroll
        for (int kc = 0; kc < 2; ++kc) {
            v16h f;
            #pragma unroll
            for (int j = 0; j < 16; ++j) {
                const int k = kc * 32 + koff + j;
                f[j] = (_Float16)px[(size_t)k * 4096];
            }
            afrag[rt][kc] = f;
        }
    }

    float best[2][8];
    int   bidx[2][8];
    #pragma unroll
    for (int rt = 0; rt < 2; ++rt)
        #pragma unroll
        for (int j = 0; j < 8; ++j) { best[rt][j] = 3.4e38f; bidx[rt][j] = 0; }

    for (int n0 = 0; n0 < NE; n0 += 16) {
        // B fragments from LDS, contiguous k = 16*h + s, same as A (lane = column)
        const _Float16* col = eL + (n0 + l16) * C_DIM + (hi ? 16 : 0);
        v16h b0 = *(const v16h*)(col);        // K =  0..31 half
        v16h b1 = *(const v16h*)(col + 32);   // K = 32..63 half

        v8f acc0 = {};
        acc0 = __builtin_amdgcn_wmma_f32_16x16x32_f16(false, afrag[0][0], false, b0,
                                                      (short)0, acc0, false, false);
        acc0 = __builtin_amdgcn_wmma_f32_16x16x32_f16(false, afrag[0][1], false, b1,
                                                      (short)0, acc0, false, false);
        v8f acc1 = {};
        acc1 = __builtin_amdgcn_wmma_f32_16x16x32_f16(false, afrag[1][0], false, b0,
                                                      (short)0, acc1, false, false);
        acc1 = __builtin_amdgcn_wmma_f32_16x16x32_f16(false, afrag[1][1], false, b1,
                                                      (short)0, acc1, false, false);

        const float nn   = nrm[n0 + l16];     // column = lane&15 in C layout
        const int   nidx = n0 + l16;
        #pragma unroll
        for (int j = 0; j < 8; ++j) {
            const float d0 = nn - 2.f * acc0[j];     // ||e||^2 - 2 z.e  (||z||^2 const)
            if (d0 < best[0][j]) { best[0][j] = d0; bidx[0][j] = nidx; }
            const float d1 = nn - 2.f * acc1[j];
            if (d1 < best[1][j]) { best[1][j] = d1; bidx[1][j] = nidx; }
        }
    }

    // cross-lane argmin within each 16-lane column group (rows 0-7 in lanes 0-15,
    // rows 8-15 in lanes 16-31; xor offsets < 16 stay inside each group)
    #pragma unroll
    for (int rt = 0; rt < 2; ++rt) {
        #pragma unroll
        for (int j = 0; j < 8; ++j) {
            float v  = best[rt][j];
            int   ix = bidx[rt][j];
            #pragma unroll
            for (int off = 1; off < 16; off <<= 1) {
                const float ov = __shfl_xor(v,  off, 32);
                const int   oi = __shfl_xor(ix, off, 32);
                if (ov < v || (ov == v && oi < ix)) { v = ov; ix = oi; }
            }
            if (l16 == 0)
                idxL[wave * 32 + rt * 16 + (hi ? 8 : 0) + j] = ix;
        }
    }
    __syncthreads();

    // ---------------- gather e[idx] and scatter to [B,C,H,W] ----------------
    // thread t handles 4 consecutive pixels (float4 stores) over 16 channels
    {
        const int m4    = (t & 63) * 4;           // local pixel group 0..252
        const int cbase = t >> 6;                 // 0..3
        const int i0 = idxL[m4 + 0];
        const int i1 = idxL[m4 + 1];
        const int i2 = idxL[m4 + 2];
        const int i3 = idxL[m4 + 3];
        const int p0 = blockIdx.x * PIX_WG + m4;
        const size_t obase = (size_t)(p0 >> 12) * 262144 + (p0 & 4095);
        #pragma unroll
        for (int k = 0; k < 16; ++k) {
            const int c = cbase + k * 4;
            float4 vres;
            vres.x = e[i0 * C_DIM + c];
            vres.y = e[i1 * C_DIM + c];
            vres.z = e[i2 * C_DIM + c];
            vres.w = e[i3 * C_DIM + c];
            *(float4*)(out + obase + (size_t)c * 4096) = vres;
        }
    }
}

extern "C" void kernel_launch(void* const* d_in, const int* in_sizes, int n_in,
                              void* d_out, int out_size, void* d_ws, size_t ws_size,
                              hipStream_t stream) {
    const float* x = (const float*)d_in[0];   // 64*64*64*64 f32
    const float* e = (const float*)d_in[1];   // 512*64 f32
    float* out = (float*)d_out;               // 64*64*64*64 f32
    (void)in_sizes; (void)n_in; (void)out_size; (void)d_ws; (void)ws_size;
    vq_argmin_gather<<<NBLK, 256, 0, stream>>>(x, e, out);
}
